// MultiHeadAttention_23149873725998
// MI455X (gfx1250) — compile-verified, matrix operand order checked by reading, not run on hardware
//
#include <hip/hip_runtime.h>
#include <hip/hip_bf16.h>
#include <stdint.h>

// ---------------------------------------------------------------------------
// MHA forward (B=4,S=1024,E=1024,H=16,D=64) for gfx1250 (CDNA5, wave32).
//  * all matmuls: v_wmma_f32_16x16x32_bf16
//  * GEMM tiles DMA'd into LDS by the Tensor Data Mover (tensor_load_to_lds)
//    with D# padding producing the conflict-free [.][40] bf16 LDS layout,
//    double-buffered via TENSORcnt.
//  * fp32->bf16 conversion done once up front (activations row-major, weights
//    transposed to [N][K] so B-tiles are row-contiguous for the TDM).
// d_out = [attn_output 4M f32 | attn_weights 64M f32]
// ws    = Xq,Xk,Xv (8MB ea) | Wtq,Wtk,Wtv,Wto (2MB ea) | Qh,Kh,Vt (8MB ea)
//         | ctx bf16 (8MB)  => 64 MB
// ---------------------------------------------------------------------------

typedef __bf16 bf16_t;
typedef __attribute__((ext_vector_type(16))) __bf16 v16bf;
typedef __attribute__((ext_vector_type(8)))  __bf16 v8bf;
typedef __attribute__((ext_vector_type(8)))  float  v8f;
typedef __attribute__((ext_vector_type(4)))  float  v4f;
typedef __attribute__((ext_vector_type(4)))  unsigned int u32x4;
typedef __attribute__((ext_vector_type(4)))  int    i32x4;
typedef __attribute__((ext_vector_type(8)))  int    i32x8;

#define B_  4
#define S_  1024
#define E_  1024
#define H_  16
#define D_  64
#define M_  (B_ * S_)

__device__ __forceinline__ v8f wmma_bf16(v16bf a, v16bf b, v8f c) {
  return __builtin_amdgcn_wmma_f32_16x16x32_bf16(
      false, a, false, b, (short)0, c, false, false);
}

// fragment element e <-> k = (e/8)*16 + (lane/16)*8 + e%8  (two 16B chunks)
__device__ __forceinline__ v16bf make_frag(const bf16_t* p0, const bf16_t* p1) {
  v8bf lo = *(const v8bf*)p0;
  v8bf hi = *(const v8bf*)p1;
  v16bf r;
#pragma unroll
  for (int i = 0; i < 8; ++i) { r[i] = lo[i]; r[i + 8] = hi[i]; }
  return r;
}

// ---------------------------------------------------------------------------
// TDM: DMA a [tile_rows x 32] bf16 tile (row pitch 1024 elems) from global
// into LDS, inserting 4 dwords (8 bf16) of padding after each 16-dword row
// so the LDS image is  tile[tile_rows][40].   D# layout per ISA ch.8.
// This toolchain's builtin takes 6 args (extra int32x8 group before cpol).
// ---------------------------------------------------------------------------
__device__ __forceinline__ void tdm_load_tile32(unsigned lds_byte_addr,
                                                const bf16_t* gp,
                                                unsigned tensor_rows,
                                                unsigned tile_rows) {
  unsigned long long ga = (unsigned long long)(uintptr_t)gp;
  u32x4 g0;
  g0[0] = 1u;                                          // count=1 (valid D#)
  g0[1] = lds_byte_addr;                               // lds_addr
  g0[2] = (unsigned)ga;                                // global_addr[31:0]
  g0[3] = (unsigned)((ga >> 32) & 0x1FFFFFFu) | (2u << 30);  // ga[56:32]|type=2
  i32x8 g1;
  g1[0] = (int)((1u << 16) |     // data_size = 2 bytes (bf16)
                (1u << 20) |     // pad_enable
                (3u << 22) |     // pad_interval: 16 dwords (one 64B row)
                (3u << 25));     // pad_amount  : 4 dwords (8 bf16)
  g1[1] = (int)(1024u << 16);                      // tensor_dim0[15:0]=1024
  g1[2] = (int)(tensor_rows << 16);                // dim0[31:16]=0 | dim1 lo
  g1[3] = (int)((tensor_rows >> 16) | (32u << 16));// dim1 hi | tile_dim0=32
  g1[4] = (int)tile_rows;                          // tile_dim1 | tile_dim2=0
  g1[5] = 1024;                                    // tensor_dim0_stride
  g1[6] = 0;
  g1[7] = 0;
  i32x4 z4 = {0, 0, 0, 0};
  i32x8 z8 = {0, 0, 0, 0, 0, 0, 0, 0};
  __builtin_amdgcn_tensor_load_to_lds(g0, g1, z4, z4, z8, 0);
}

// ---------------------------------------------------------------------------
// Converters (one-shot preprocessing)
// ---------------------------------------------------------------------------
__global__ __launch_bounds__(256) void convert_bf16(const float* __restrict__ src,
                                                    bf16_t* __restrict__ dst) {
  const size_t i = ((size_t)blockIdx.x * 256 + threadIdx.x) * 8;
  v4f a = *(const v4f*)(src + i);
  v4f b = *(const v4f*)(src + i + 4);
  v8bf o;
#pragma unroll
  for (int j = 0; j < 4; ++j) { o[j] = (bf16_t)a[j]; o[j + 4] = (bf16_t)b[j]; }
  *(v8bf*)(dst + i) = o;
}

// W[K][N] f32 -> Wt[N][K] bf16  (64x64 LDS tiles, coalesced both sides)
__global__ __launch_bounds__(256) void convert_w_t(const float* __restrict__ W,
                                                   bf16_t* __restrict__ Wt) {
  __shared__ float t[64][65];
  const int tid = threadIdx.x;
  const int k0 = blockIdx.y * 64, n0 = blockIdx.x * 64;
#pragma unroll
  for (int i = 0; i < 16; ++i) {
    const int idx = i * 256 + tid, r = idx >> 6, c = idx & 63;
    t[r][c] = W[(size_t)(k0 + r) * E_ + n0 + c];
  }
  __syncthreads();
#pragma unroll
  for (int i = 0; i < 16; ++i) {
    const int idx = i * 256 + tid, r = idx >> 6, c = idx & 63;
    Wt[(size_t)(n0 + r) * E_ + k0 + c] = (bf16_t)t[c][r];
  }
}

// ---------------------------------------------------------------------------
// GEMM: out = A[M,1024](bf16) @ Wt[N,K](bf16, pre-transposed) + bias
// MODE 0: out bf16 -> [B,H,S,D]   MODE 1: out bf16 -> [B,H,D,S]
// MODE 2: out f32  -> [M,E] (attn_output)
// BM=128,BN=64,BK=32; 8 waves (4x2), 2x2 WMMA tiles each; TDM double-buffer.
// ---------------------------------------------------------------------------
template <int MODE>
__global__ __launch_bounds__(256) void gemm_tdm(const bf16_t* __restrict__ A,
                                                const bf16_t* __restrict__ Wt,
                                                const float* __restrict__ bias,
                                                void* __restrict__ out_) {
  __shared__ __align__(128) bf16_t As[2][128][40];
  __shared__ __align__(128) bf16_t Bs[2][64][40];

  const int tid  = threadIdx.x;
  const int wave = tid >> 5;
  const int lane = tid & 31;
  const int wm = wave & 3, wn = wave >> 2;
  const int hi = lane >> 4, ln = lane & 15;
  const int block_n = blockIdx.x * 64;
  const int block_m = blockIdx.y * 128;

  const bf16_t* Abase = A + (size_t)block_m * E_;
  const bf16_t* Bbase = Wt + (size_t)block_n * E_;

  v8f acc[2][2];
#pragma unroll
  for (int i = 0; i < 2; ++i)
#pragma unroll
    for (int j = 0; j < 2; ++j)
#pragma unroll
      for (int r = 0; r < 8; ++r) acc[i][j][r] = 0.0f;

  if (wave == 0) {  // prologue: buffer 0 <- k=0
    tdm_load_tile32((unsigned)(uintptr_t)&As[0][0][0], Abase, M_, 128);
    tdm_load_tile32((unsigned)(uintptr_t)&Bs[0][0][0], Bbase, E_, 64);
  }

  for (int k0 = 0; k0 < E_; k0 += 32) {
    const int cur = (k0 >> 5) & 1;
    if (wave == 0) {
      if (k0 + 32 < E_) {  // prefetch next pair into the other buffer
        tdm_load_tile32((unsigned)(uintptr_t)&As[cur ^ 1][0][0],
                        Abase + (k0 + 32), M_, 128);
        tdm_load_tile32((unsigned)(uintptr_t)&Bs[cur ^ 1][0][0],
                        Bbase + (k0 + 32), E_, 64);
        __builtin_amdgcn_s_wait_tensorcnt(2);  // in-order: current pair done
      } else {
        __builtin_amdgcn_s_wait_tensorcnt(0);
      }
    }
    __syncthreads();

    const int hk = hi * 8;
    v16bf af[2], bfr[2];
#pragma unroll
    for (int mi = 0; mi < 2; ++mi) {
      const int row = wm * 32 + mi * 16 + ln;
      af[mi] = make_frag(&As[cur][row][hk], &As[cur][row][hk + 16]);
    }
#pragma unroll
    for (int ni = 0; ni < 2; ++ni) {
      const int col = wn * 32 + ni * 16 + ln;
      bfr[ni] = make_frag(&Bs[cur][col][hk], &Bs[cur][col][hk + 16]);
    }
#pragma unroll
    for (int mi = 0; mi < 2; ++mi)
#pragma unroll
      for (int ni = 0; ni < 2; ++ni)
        acc[mi][ni] = wmma_bf16(af[mi], bfr[ni], acc[mi][ni]);
    __syncthreads();  // safe to overwrite 'cur' next iteration
  }

#pragma unroll
  for (int mi = 0; mi < 2; ++mi) {
#pragma unroll
    for (int ni = 0; ni < 2; ++ni) {
      const int n = block_n + wn * 32 + ni * 16 + ln;
      const float bv = bias[n];
#pragma unroll
      for (int r = 0; r < 8; ++r) {
        const int m = block_m + wm * 32 + mi * 16 + r + hi * 8;
        const float v = acc[mi][ni][r] + bv;
        if (MODE == 2) {
          ((float*)out_)[(size_t)m * E_ + n] = v;
        } else {
          const int b = m >> 10, s = m & 1023;
          const int h = n >> 6,  d = n & 63;
          size_t idx;
          if (MODE == 0) idx = (((size_t)(b * H_ + h)) * S_ + s) * D_ + d;
          else           idx = (((size_t)(b * H_ + h)) * D_ + d) * S_ + s;
          ((bf16_t*)out_)[idx] = (bf16_t)v;
        }
      }
    }
  }
}

// ---------------------------------------------------------------------------
// Attention: workgroup = (b,h) x 16-row Q tile vs all 1024 keys; 8 waves each
// own a 128-key chunk.  Scores in WMMA accum registers; softmax via shfl_xor
// + LDS cross-wave reduce; mask staged to LDS with 16B loads; weights f32 to
// d_out; probs bf16 to LDS for P@V; ctx written bf16.
// ---------------------------------------------------------------------------
__global__ __launch_bounds__(256) void attention_kernel(
    const bf16_t* __restrict__ Qh, const bf16_t* __restrict__ Kh,
    const bf16_t* __restrict__ Vt, const unsigned char* __restrict__ mask,
    float* __restrict__ weights, bf16_t* __restrict__ ctx) {
  __shared__ bf16_t probs[16][1032];
  __shared__ union {                      // lifetimes are disjoint
    unsigned char m[16][1024];            // mask slab (phase 1)
    float part[4][16][64];                // PV reduction (phase 2)
  } uu;
  __shared__ float redbuf[8][16];
  __shared__ float rowstat[16];

  const int tid  = threadIdx.x;
  const int wave = tid >> 5;
  const int lane = tid & 31;
  const int hi = lane >> 4, ln = lane & 15;

  const int qt = blockIdx.x;
  const int bh = blockIdx.y;
  const int b = bh >> 4, h = bh & 15;

  const bf16_t* Qp = Qh + ((size_t)bh * S_ + qt * 16) * D_;
  const bf16_t* Kp = Kh + (size_t)bh * S_ * D_;
  const bf16_t* Vp = Vt + (size_t)bh * D_ * S_;

  // stage mask slab [16][1024] bytes with 16B vector loads (64B/thread)
  {
    const u32x4* msrc =
        (const u32x4*)(mask + ((size_t)b * S_ + qt * 16) * S_);
    u32x4* mdst = (u32x4*)uu.m;
#pragma unroll
    for (int i = 0; i < 4; ++i) {
      const int c = i * 256 + tid;   // 1024 chunks of 16B
      mdst[c] = msrc[c];
    }
  }

  v16bf aq[2];
#pragma unroll
  for (int ks = 0; ks < 2; ++ks) {
    const int d0 = ks * 32 + hi * 8;
    const bf16_t* qp = Qp + (size_t)ln * D_ + d0;
    aq[ks] = make_frag(qp, qp + 16);
  }

  const int jbase = wave * 128;
  v8f acc[8];
#pragma unroll
  for (int t = 0; t < 8; ++t)
#pragma unroll
    for (int r = 0; r < 8; ++r) acc[t][r] = 0.0f;

#pragma unroll
  for (int nt = 0; nt < 8; ++nt) {
    const int j = jbase + nt * 16 + ln;
    const bf16_t* kp = Kp + (size_t)j * D_;
#pragma unroll
    for (int ks = 0; ks < 2; ++ks) {
      const int d0 = ks * 32 + hi * 8;
      acc[nt] = wmma_bf16(aq[ks], make_frag(kp + d0, kp + d0 + 16), acc[nt]);
    }
  }
  __syncthreads();  // mask slab visible

  float m8[8];
#pragma unroll
  for (int r = 0; r < 8; ++r) m8[r] = -3.0e38f;
#pragma unroll
  for (int nt = 0; nt < 8; ++nt) {
    const int j = jbase + nt * 16 + ln;
#pragma unroll
    for (int r = 0; r < 8; ++r) {
      float v = acc[nt][r] * 0.125f;                  // 1/sqrt(64)
      v = uu.m[r + hi * 8][j] ? v : -1.0e9f;
      acc[nt][r] = v;
      m8[r] = fmaxf(m8[r], v);
    }
  }
#pragma unroll
  for (int off = 8; off >= 1; off >>= 1)
#pragma unroll
    for (int r = 0; r < 8; ++r)
      m8[r] = fmaxf(m8[r], __shfl_xor(m8[r], off, 32));
  if (ln == 0) {
#pragma unroll
    for (int r = 0; r < 8; ++r) redbuf[wave][r + hi * 8] = m8[r];
  }
  __syncthreads();
  if (tid < 16) {
    float mv = redbuf[0][tid];
#pragma unroll
    for (int w = 1; w < 8; ++w) mv = fmaxf(mv, redbuf[w][tid]);
    rowstat[tid] = mv;
  }
  __syncthreads();
  float rmax[8];
#pragma unroll
  for (int r = 0; r < 8; ++r) rmax[r] = rowstat[r + hi * 8];
  __syncthreads();

  float s8[8];
#pragma unroll
  for (int r = 0; r < 8; ++r) s8[r] = 0.0f;
#pragma unroll
  for (int nt = 0; nt < 8; ++nt)
#pragma unroll
    for (int r = 0; r < 8; ++r) {
      const float e = __expf(acc[nt][r] - rmax[r]);
      acc[nt][r] = e;
      s8[r] += e;
    }
#pragma unroll
  for (int off = 8; off >= 1; off >>= 1)
#pragma unroll
    for (int r = 0; r < 8; ++r) s8[r] += __shfl_xor(s8[r], off, 32);
  if (ln == 0) {
#pragma unroll
    for (int r = 0; r < 8; ++r) redbuf[wave][r + hi * 8] = s8[r];
  }
  __syncthreads();
  if (tid < 16) {
    float sv = redbuf[0][tid];
#pragma unroll
    for (int w = 1; w < 8; ++w) sv += redbuf[w][tid];
    rowstat[tid] = 1.0f / sv;
  }
  __syncthreads();
  float rinv[8];
#pragma unroll
  for (int r = 0; r < 8; ++r) rinv[r] = rowstat[r + hi * 8];

  float* wout = weights + ((size_t)bh * S_ + qt * 16) * S_;
#pragma unroll
  for (int nt = 0; nt < 8; ++nt) {
    const int j = jbase + nt * 16 + ln;
#pragma unroll
    for (int r = 0; r < 8; ++r) {
      const int qr = r + hi * 8;
      const float p = acc[nt][r] * rinv[r];
      wout[(size_t)qr * S_ + j] = p;
      probs[qr][j] = (bf16_t)p;
    }
  }
  __syncthreads();  // probs ready; mask slab dead -> uu.part reusable

  v8f accv[4];
#pragma unroll
  for (int t = 0; t < 4; ++t)
#pragma unroll
    for (int r = 0; r < 8; ++r) accv[t][r] = 0.0f;
#pragma unroll
  for (int ks = 0; ks < 4; ++ks) {
    const int j0 = jbase + ks * 32 + hi * 8;
    v16bf ap = make_frag(&probs[ln][j0], &probs[ln][j0 + 16]);
#pragma unroll
    for (int nt = 0; nt < 4; ++nt) {
      const int d = nt * 16 + ln;
      const bf16_t* vp = Vp + (size_t)d * S_ + j0;  // Vt [D,S]: contiguous
      accv[nt] = wmma_bf16(ap, make_frag(vp, vp + 16), accv[nt]);
    }
  }

  if (wave >= 4) {
#pragma unroll
    for (int nt = 0; nt < 4; ++nt)
#pragma unroll
      for (int r = 0; r < 8; ++r)
        uu.part[wave - 4][r + hi * 8][nt * 16 + ln] = accv[nt][r];
  }
  __syncthreads();
  if (wave < 4) {
#pragma unroll
    for (int nt = 0; nt < 4; ++nt)
#pragma unroll
      for (int r = 0; r < 8; ++r)
        uu.part[wave][r + hi * 8][nt * 16 + ln] += accv[nt][r];
  }
  __syncthreads();
#pragma unroll
  for (int i = 0; i < 4; ++i) {
    const int e = i * 256 + tid;
    const int qr = e >> 6, d = e & 63;
    const float v = uu.part[0][qr][d] + uu.part[1][qr][d] +
                    uu.part[2][qr][d] + uu.part[3][qr][d];
    ctx[((size_t)b * S_ + qt * 16 + qr) * E_ + h * D_ + d] = (bf16_t)v;
  }
}

// ---------------------------------------------------------------------------
extern "C" void kernel_launch(void* const* d_in, const int* in_sizes, int n_in,
                              void* d_out, int out_size, void* d_ws, size_t ws_size,
                              hipStream_t stream) {
  const float* query = (const float*)d_in[0];
  const float* key_  = (const float*)d_in[1];
  const float* value = (const float*)d_in[2];
  const unsigned char* mask = (const unsigned char*)d_in[3];
  const float* Wq = (const float*)d_in[4];
  const float* bq = (const float*)d_in[5];
  const float* Wk = (const float*)d_in[6];
  const float* bk = (const float*)d_in[7];
  const float* Wv = (const float*)d_in[8];
  const float* bv = (const float*)d_in[9];
  const float* Wo = (const float*)d_in[10];
  const float* bo = (const float*)d_in[11];

  const size_t XB = (size_t)M_ * E_ * sizeof(bf16_t);          // 8 MB
  const size_t WB = (size_t)E_ * E_ * sizeof(bf16_t);          // 2 MB
  uint8_t* w = (uint8_t*)d_ws;
  bf16_t* Xq  = (bf16_t*)(w);
  bf16_t* Xk  = (bf16_t*)(w + XB);
  bf16_t* Xv  = (bf16_t*)(w + 2 * XB);
  bf16_t* Wtq = (bf16_t*)(w + 3 * XB);
  bf16_t* Wtk = (bf16_t*)(w + 3 * XB + WB);
  bf16_t* Wtv = (bf16_t*)(w + 3 * XB + 2 * WB);
  bf16_t* Wto = (bf16_t*)(w + 3 * XB + 3 * WB);
  bf16_t* Qh  = (bf16_t*)(w + 3 * XB + 4 * WB);
  bf16_t* Kh  = (bf16_t*)(w + 4 * XB + 4 * WB);
  bf16_t* Vt  = (bf16_t*)(w + 5 * XB + 4 * WB);
  bf16_t* ctx = (bf16_t*)(w + 6 * XB + 4 * WB);

  float* attn_out = (float*)d_out;
  float* attn_w   = (float*)d_out + (size_t)B_ * S_ * E_;

  const int cvt_blocks = (M_ * E_) / (256 * 8);  // 2048
  convert_bf16<<<cvt_blocks, 256, 0, stream>>>(query, Xq);
  convert_bf16<<<cvt_blocks, 256, 0, stream>>>(key_,  Xk);
  convert_bf16<<<cvt_blocks, 256, 0, stream>>>(value, Xv);
  const dim3 wt_grid(16, 16);
  convert_w_t<<<wt_grid, 256, 0, stream>>>(Wq, Wtq);
  convert_w_t<<<wt_grid, 256, 0, stream>>>(Wk, Wtk);
  convert_w_t<<<wt_grid, 256, 0, stream>>>(Wv, Wtv);
  convert_w_t<<<wt_grid, 256, 0, stream>>>(Wo, Wto);

  const dim3 gemm_grid(E_ / 64, M_ / 128);
  gemm_tdm<0><<<gemm_grid, 256, 0, stream>>>(Xq, Wtq, bq, Qh);
  gemm_tdm<0><<<gemm_grid, 256, 0, stream>>>(Xk, Wtk, bk, Kh);
  gemm_tdm<1><<<gemm_grid, 256, 0, stream>>>(Xv, Wtv, bv, Vt);

  attention_kernel<<<dim3(S_ / 16, B_ * H_), 256, 0, stream>>>(
      Qh, Kh, Vt, mask, attn_w, ctx);

  gemm_tdm<2><<<gemm_grid, 256, 0, stream>>>(ctx, Wto, bo, attn_out);
}
